// SE3TransformerConvV2_23313082483618
// MI455X (gfx1250) — matrix operand order repaired and checked by reading, not compiled, not run on hardware
//
#include <hip/hip_runtime.h>
#include <cmath>
#include <complex>
#include <vector>

typedef _Float16 f16;
typedef __attribute__((ext_vector_type(16))) _Float16 v16h;
typedef __attribute__((ext_vector_type(8)))  _Float16 v8h;
typedef __attribute__((ext_vector_type(8)))  float    v8f;

#define LDSW 356   // LDS row pitch (floats): mult of 4 (float4) and 36m%64 distinct

// ---------------------------------------------------------------------------
// WMMA helper: D = A(16x32 f16) * B(32x16 f16) + C(16x16 f32)
// ---------------------------------------------------------------------------
__device__ __forceinline__ v8f wmma_f16(v16h a, v16h b, v8f c) {
    return __builtin_amdgcn_wmma_f32_16x16x32_f16(false, a, false, b,
                                                  (short)0, c, false, false);
}

// A-fragment k index for value t (0..15), half-lane h (lane>>4)
__device__ __forceinline__ int a_frag_k(int t, int h) {
    return (h << 4) | t;
}

// ---------------------------------------------------------------------------
// f16 GEMM: C[M,O] = act(A16[:, acol0:acol0+K] @ W16[O,K]^T + bias)
// one wave per 16x16 tile; all fragment loads are contiguous vector loads.
// K must be a multiple of 32. C (f32) and C16 (f16) each optional.
// ---------------------------------------------------------------------------
__global__ __launch_bounds__(32)
void gemm16_kernel(const f16* __restrict__ A, int lda, int acol0,
                   const f16* __restrict__ W, const float* __restrict__ bias,
                   float* __restrict__ C, f16* __restrict__ C16, int ldc,
                   int M, int K, int act)
{
    const int lane = threadIdx.x;
    const int m0 = blockIdx.x * 16;
    const int n0 = blockIdx.y * 16;
    const int h  = lane >> 4;
    const int l15 = lane & 15;
    int m = m0 + l15; if (m > M - 1) m = M - 1;
    const int n = n0 + l15;

    const f16* __restrict__ arow = A + (long)m * lda + acol0 + h * 16;
    const f16* __restrict__ wrow = W + (long)n * K + h * 16;

    v8f acc = {};
    for (int kk = 0; kk < K; kk += 32) {
        v8h a0 = *(const v8h*)(arow + kk);        // K = h*16 .. h*16+7
        v8h a1 = *(const v8h*)(arow + kk + 8);    // K = h*16 + 8 ..
        v16h av = __builtin_shufflevector(a0, a1, 0,1,2,3,4,5,6,7,
                                                  8,9,10,11,12,13,14,15);
        v16h bv = *(const v16h*)(wrow + kk);      // K = h*16 + 0..15
        acc = wmma_f16(av, bv, acc);
    }
#pragma unroll
    for (int r = 0; r < 8; ++r) {
        int row = m0 + h * 8 + r;
        if (row < M) {
            float x = acc[r] + bias[n];
            if (act == 1) x = x / (1.f + __expf(-x));   // silu
            if (C)   C[(long)row * ldc + n] = x;
            if (C16) C16[(long)row * ldc + n] = (f16)x;
        }
    }
}

// ---------------------------------------------------------------------------
// prep: f32 -> f16 convert; tpw[K,MO] f32 -> [MO,K] f16 transpose
// ---------------------------------------------------------------------------
__global__ void cvt_f16_kernel(const float* __restrict__ in, f16* __restrict__ out, long n)
{
    long i = (long)blockIdx.x * blockDim.x + threadIdx.x;
    if (i < n) out[i] = (f16)in[i];
}

__global__ void tpw_t_kernel(const float* __restrict__ in, f16* __restrict__ out,
                             int K, int MO)
{
    long i = (long)blockIdx.x * blockDim.x + threadIdx.x;
    if (i >= (long)K * MO) return;
    int k = (int)(i / MO), w = (int)(i % MO);
    out[(long)w * K + k] = (f16)in[i];
}

// ---------------------------------------------------------------------------
// rbf filter + x_src (K=20 -> VALU), emits f16 for downstream WMMA GEMMs
// ---------------------------------------------------------------------------
__global__ void rbf_xsrc_kernel(const float* __restrict__ xs, const float* __restrict__ rbf,
                                const float* __restrict__ env,
                                const float* __restrict__ W, const float* __restrict__ b,
                                f16* __restrict__ x_src16, int N)
{
    int i = blockIdx.x * blockDim.x + threadIdx.x;
    if (i >= N * 128) return;
    int n = i >> 7, c = i & 127;
    const float* r = rbf + (long)n * 20;
    const float* w = W + (long)c * 20;
    float a = 0.f;
#pragma unroll
    for (int t = 0; t < 20; ++t) a += r[t] * w[t];
    a = (a + b[c]) * env[n];
    x_src16[i] = (f16)(xs[i] * a);
}

// ---------------------------------------------------------------------------
// elementwise tensor product gating: mixed = xv*s2[irrep] + rsh*s1[irrep]
// ---------------------------------------------------------------------------
__global__ void etp_kernel(const float* __restrict__ xv, const float* __restrict__ rsh,
                           const float* __restrict__ s1, const float* __restrict__ s2,
                           float* __restrict__ mixed, int N)
{
    int i = blockIdx.x * blockDim.x + threadIdx.x;
    if (i >= N * 352) return;
    int n = i / 352, c = i % 352;
    int u = (c < 192) ? (c / 3) : (64 + (c - 192) / 5);
    mixed[i] = xv[i] * s2[(long)n * 96 + u] + rsh[i] * s1[(long)n * 96 + u];
}

// ---------------------------------------------------------------------------
// attention: logits from f16 q/k + segment max (branch float atomic-max trick)
// ---------------------------------------------------------------------------
__device__ __forceinline__ void atomicMaxFloat(float* addr, float v) {
    if (v >= 0.f) atomicMax((int*)addr, __float_as_int(v));
    else          atomicMin((unsigned int*)addr, __float_as_uint(v));
}

__global__ void att_logit_kernel(const f16* __restrict__ q, const f16* __restrict__ k,
                                 const int* __restrict__ ei,
                                 float* __restrict__ logits, float* __restrict__ mx, int E)
{
    int i = blockIdx.x * blockDim.x + threadIdx.x;
    if (i >= E * 8) return;
    int e = i >> 3, hh = i & 7;
    int s = ei[e], d = ei[E + e];
    const v8h* qp = (const v8h*)(q + (long)d * 256 + hh * 32);
    const v8h* kp = (const v8h*)(k + (long)s * 256 + hh * 32);
    float acc = 0.f;
#pragma unroll
    for (int t = 0; t < 4; ++t) {
        v8h a = qp[t], b = kp[t];
#pragma unroll
        for (int j = 0; j < 8; ++j) acc += (float)a[j] * (float)b[j];
    }
    acc *= 0.17677669529663687f;          // 1/sqrt(32)
    logits[i] = acc;
    atomicMaxFloat(&mx[(long)d * 8 + hh], acc);
}

__global__ void att_exp_kernel(const int* __restrict__ ei, float* __restrict__ logits,
                               const float* __restrict__ mx, float* __restrict__ den, int E)
{
    int i = blockIdx.x * blockDim.x + threadIdx.x;
    if (i >= E * 8) return;
    int e = i >> 3, hh = i & 7;
    int d = ei[E + e];
    float ex = __expf(logits[i] - mx[(long)d * 8 + hh]);
    logits[i] = ex;
    atomicAdd(&den[(long)d * 8 + hh], ex);
}

__global__ void scatter_scalar_kernel(const int* __restrict__ ei, const float* __restrict__ e_att,
                                      const float* __restrict__ den, const float* __restrict__ v,
                                      float* __restrict__ out, int E)
{
    int i = blockIdx.x * blockDim.x + threadIdx.x;
    if (i >= E * 8) return;
    int e = i >> 3, hh = i & 7;
    int s = ei[e], d = ei[E + e];
    float w = e_att[i] / (den[(long)d * 8 + hh] + 1e-16f);
    const float* vp = v + (long)s * 384 + 256 + hh * 16;   // val = v[:,256:384]
    float* op = out + (long)d * 128 + hh * 16;
#pragma unroll
    for (int t = 0; t < 16; ++t) atomicAdd(&op[t], w * vp[t]);
}

__global__ void scatter_agg_kernel(const int* __restrict__ ei, const float* __restrict__ mixed,
                                   float* __restrict__ agg, int E)
{
    long i = (long)blockIdx.x * blockDim.x + threadIdx.x;
    if (i >= (long)E * 88) return;          // 352/4 float4 per edge
    int e = (int)(i / 88), c = (int)(i % 88) * 4;
    float4 mv = *(const float4*)(mixed + (long)ei[e] * 352 + c);
    float* ap = agg + (long)ei[E + e] * 352 + c;
    atomicAdd(ap + 0, mv.x); atomicAdd(ap + 1, mv.y);
    atomicAdd(ap + 2, mv.z); atomicAdd(ap + 3, mv.w);
}

__global__ void initf_kernel(float* __restrict__ p, float v, long n)
{
    long i = (long)blockIdx.x * blockDim.x + threadIdx.x;
    if (i < n) p[i] = v;
}

__global__ void copyf_kernel(float* __restrict__ d, const float* __restrict__ s, long n)
{
    long i = (long)blockIdx.x * blockDim.x + threadIdx.x;
    if (i < n) d[i] = s[i];
}

// ---------------------------------------------------------------------------
// Fully-connected tensor product, WMMA formulation:
//   out_k[node, w] += sum_uv xy_k[node,(u,v)] * tpw[(u,v), w]
// xy_k computed on the fly from LDS-staged x(agg)/y(xvec) tiles; B is the
// pre-transposed f16 weight [MO][K] -> one v16h load per fragment.
// ---------------------------------------------------------------------------
template<int M1, int D1, int XO, int M2, int D2, int YO, int CGO,
         int KO, int MO, int NWT>
__device__ void fctp_path(const float* xs, const float* ys, const float* cgs,
                          const f16* __restrict__ twT, v8f (&acc)[KO * NWT], int lane)
{
    const int m = lane & 15, h = lane >> 4;
    constexpr int K = M1 * M2;
    const f16* __restrict__ twb = twT + (long)m * K + h * 16;   // col = wt*16 + m
    for (int kk = 0; kk < K; kk += 32) {
        v16h bf[NWT];
#pragma unroll
        for (int wt = 0; wt < NWT; ++wt)
            bf[wt] = *(const v16h*)(twb + (long)wt * 16 * K + kk);
        v16h af[KO];
#pragma unroll
        for (int t = 0; t < 16; ++t) {
            int uv = kk + a_frag_k(t, h);
            int u = uv / M2, v = uv % M2;
            float xr[D1], yr[D2];
#pragma unroll
            for (int i = 0; i < D1; ++i) xr[i] = xs[m * LDSW + XO + u * D1 + i];
#pragma unroll
            for (int j = 0; j < D2; ++j) yr[j] = ys[m * LDSW + YO + v * D2 + j];
#pragma unroll
            for (int k = 0; k < KO; ++k) {
                float a = 0.f;
#pragma unroll
                for (int i = 0; i < D1; ++i)
#pragma unroll
                    for (int j = 0; j < D2; ++j)
                        a += cgs[CGO + (i * D2 + j) * KO + k] * xr[i] * yr[j];
                af[k][t] = (f16)a;
            }
        }
#pragma unroll
        for (int k = 0; k < KO; ++k)
#pragma unroll
            for (int wt = 0; wt < NWT; ++wt)
                acc[k * NWT + wt] = wmma_f16(af[k], bf[wt], acc[k * NWT + wt]);
    }
}

template<int IO>
__global__ __launch_bounds__(32)
void fctp_kernel(const float* __restrict__ agg, const float* __restrict__ xv,
                 const float* __restrict__ cgbuf,
                 const f16* __restrict__ tw0, const f16* __restrict__ tw1,
                 const f16* __restrict__ tw2, const f16* __restrict__ tw3,
                 float* __restrict__ out, int N)
{
    constexpr int KO   = (IO == 0) ? 3 : 5;
    constexpr int MO   = (IO == 0) ? 64 : 32;
    constexpr int NWT  = MO / 16;
    constexpr int OOFF = (IO == 0) ? 0 : 192;
    const float alpha  = (IO == 0) ? 0.018042195912175807f   // sqrt(3)/96
                                   : 0.023292373145817495f;  // sqrt(5)/96

    __shared__ float xs[16 * LDSW];
    __shared__ float ys[16 * LDSW];
    __shared__ float cgs[512];

    const int lane = threadIdx.x;
    const int n0 = blockIdx.x * 16;

    // float4-vectorized tile staging: 16 rows x 88 float4
    for (int idx = lane; idx < 16 * 88; idx += 32) {
        int r = idx / 88, c = (idx % 88) * 4;
        int row = n0 + r; if (row > N - 1) row = N - 1;
        float4 xa = *(const float4*)(agg + (long)row * 352 + c);
        float4 ya = *(const float4*)(xv  + (long)row * 352 + c);
        *(float4*)(xs + r * LDSW + c) = xa;
        *(float4*)(ys + r * LDSW + c) = ya;
    }
    for (int idx = lane; idx < 512; idx += 32) cgs[idx] = cgbuf[idx];
    __syncthreads();

    v8f acc[KO * NWT] = {};

    if constexpr (IO == 0) {
        fctp_path<64, 3, 0,   64, 3, 0,     0, KO, MO, NWT>(xs, ys, cgs, tw0, acc, lane);
        fctp_path<64, 3, 0,   32, 5, 192,  27, KO, MO, NWT>(xs, ys, cgs, tw1, acc, lane);
        fctp_path<32, 5, 192, 64, 3, 0,    72, KO, MO, NWT>(xs, ys, cgs, tw2, acc, lane);
        fctp_path<32, 5, 192, 32, 5, 192, 117, KO, MO, NWT>(xs, ys, cgs, tw3, acc, lane);
    } else {
        fctp_path<64, 3, 0,   64, 3, 0,   192, KO, MO, NWT>(xs, ys, cgs, tw0, acc, lane);
        fctp_path<64, 3, 0,   32, 5, 192, 237, KO, MO, NWT>(xs, ys, cgs, tw1, acc, lane);
        fctp_path<32, 5, 192, 64, 3, 0,   312, KO, MO, NWT>(xs, ys, cgs, tw2, acc, lane);
        fctp_path<32, 5, 192, 32, 5, 192, 387, KO, MO, NWT>(xs, ys, cgs, tw3, acc, lane);
    }

    const int h = lane >> 4, l15 = lane & 15;
#pragma unroll
    for (int k = 0; k < KO; ++k)
#pragma unroll
        for (int wt = 0; wt < NWT; ++wt)
#pragma unroll
            for (int r = 0; r < 8; ++r) {
                int node = n0 + h * 8 + r;
                if (node < N) {
                    int c = OOFF + (wt * 16 + l15) * KO + k;
                    long oidx = (long)N * 128 + (long)node * 352 + c;
                    out[oidx] = acc[k * NWT + wt][r] * alpha + xv[(long)node * 352 + c];
                }
            }
}

// ---------------------------------------------------------------------------
// Host: real Wigner-3j in e3nn convention (exact reference translation)
// ---------------------------------------------------------------------------
static double factd(int n) { double r = 1.0; for (int i = 2; i <= n; ++i) r *= i; return r; }

static double cgc(int j1, int m1, int j2, int m2, int j3, int m3) {
    if (m1 + m2 != m3) return 0.0;
    double pre = std::sqrt((2 * j3 + 1) * factd(j1 + j2 - j3) * factd(j1 - j2 + j3) *
                           factd(-j1 + j2 + j3) / factd(j1 + j2 + j3 + 1));
    pre *= std::sqrt(factd(j3 + m3) * factd(j3 - m3) * factd(j1 - m1) * factd(j1 + m1) *
                     factd(j2 - m2) * factd(j2 + m2));
    double s = 0.0;
    for (int k = 0; k <= j1 + j2 - j3; ++k) {
        int t0 = k, t1 = j1 + j2 - j3 - k, t2 = j1 - m1 - k, t3 = j2 + m2 - k,
            t4 = j3 - j2 + m1 + k, t5 = j3 - j1 - m2 + k;
        if (t0 < 0 || t1 < 0 || t2 < 0 || t3 < 0 || t4 < 0 || t5 < 0) continue;
        double d = factd(t0) * factd(t1) * factd(t2) * factd(t3) * factd(t4) * factd(t5);
        s += ((k & 1) ? -1.0 : 1.0) / d;
    }
    return pre * s;
}

static void qmat(int l, std::complex<double>* q) {
    using cd = std::complex<double>;
    int d = 2 * l + 1;
    for (int a = 0; a < d * d; ++a) q[a] = 0.0;
    const cd I(0.0, 1.0);
    double r2 = 1.0 / std::sqrt(2.0);
    for (int m = -l; m < 0; ++m) {
        q[(l + m) * d + (l - m)] = r2;
        q[(l + m) * d + (l + m)] = -I * r2;
    }
    q[l * d + l] = 1.0;
    for (int m = 1; m <= l; ++m) {
        double sg = (m % 2 == 0) ? 1.0 : -1.0;
        q[(l + m) * d + (l + m)] = sg * r2;
        q[(l + m) * d + (l - m)] = I * sg * r2;
    }
    cd ph(1.0, 0.0);
    for (int t = 0; t < l; ++t) ph *= cd(0.0, -1.0);
    for (int a = 0; a < d * d; ++a) q[a] *= ph;
}

static void w3j_host(int l1, int l2, int l3, float* out) {
    int d1 = 2 * l1 + 1, d2 = 2 * l2 + 1, d3 = 2 * l3 + 1;
    std::vector<std::complex<double>> C(d1 * d2 * d3, 0.0);
    for (int m1 = -l1; m1 <= l1; ++m1)
        for (int m2 = -l2; m2 <= l2; ++m2) {
            int m3 = m1 + m2;
            if (std::abs(m3) <= l3)
                C[((l1 + m1) * d2 + (l2 + m2)) * d3 + (l3 + m3)] = cgc(l1, m1, l2, m2, l3, m3);
        }
    std::vector<std::complex<double>> q1(d1 * d1), q2(d2 * d2), q3(d3 * d3);
    qmat(l1, q1.data()); qmat(l2, q2.data()); qmat(l3, q3.data());
    std::vector<std::complex<double>> C2(d1 * d2 * d3, 0.0);
    for (int j = 0; j < d1; ++j)
        for (int L = 0; L < d2; ++L)
            for (int M = 0; M < d3; ++M) {
                std::complex<double> s = 0.0;
                for (int i = 0; i < d1; ++i)
                    for (int k = 0; k < d2; ++k)
                        for (int n = 0; n < d3; ++n)
                            s += q1[i * d1 + j] * q2[k * d2 + L] * std::conj(q3[n * d3 + M]) *
                                 C[(i * d2 + k) * d3 + n];
                C2[(j * d2 + L) * d3 + M] = s;
            }
    double nr = 0.0, ni = 0.0;
    for (auto& z : C2) { nr += z.real() * z.real(); ni += z.imag() * z.imag(); }
    bool useReal = nr >= ni;
    double nrm = std::sqrt(useReal ? nr : ni);
    for (int a = 0; a < d1 * d2 * d3; ++a)
        out[a] = (float)((useReal ? C2[a].real() : C2[a].imag()) / nrm);
}

// ---------------------------------------------------------------------------
extern "C" void kernel_launch(void* const* d_in, const int* in_sizes, int n_in,
                              void* d_out, int out_size, void* d_ws, size_t ws_size,
                              hipStream_t stream)
{
    const float* x_scalar = (const float*)d_in[0];
    const float* x_vector = (const float*)d_in[1];
    const int*   ei       = (const int*)d_in[2];
    const float* rbf      = (const float*)d_in[3];
    const float* rsh      = (const float*)d_in[4];
    const float* env      = (const float*)d_in[5];
    const float* W_rbf = (const float*)d_in[6],  *b_rbf = (const float*)d_in[7];
    const float* W_q   = (const float*)d_in[8],  *b_q   = (const float*)d_in[9];
    const float* W_k   = (const float*)d_in[10], *b_k   = (const float*)d_in[11];
    const float* W_v1  = (const float*)d_in[12], *b_v1  = (const float*)d_in[13];
    const float* W_v2  = (const float*)d_in[14], *b_v2  = (const float*)d_in[15];
    const float* W_rshf= (const float*)d_in[16], *b_rshf= (const float*)d_in[17];
    const float* W_irf = (const float*)d_in[18], *b_irf = (const float*)d_in[19];
    const float* tpw[8];
    for (int p = 0; p < 8; ++p) tpw[p] = (const float*)d_in[20 + p];

    const int N = in_sizes[0] / 128;
    const int E = in_sizes[2] / 2;
    float* out = (float*)d_out;

    // ---- workspace carve (byte offsets, 64B aligned) ----
    char* base = (char*)d_ws;
    size_t off = 0;
    auto carve = [&](size_t bytes) -> char* {
        off = (off + 63) & ~(size_t)63;
        char* p = base + off;
        off += bytes;
        return p;
    };
    float* d_cg   = (float*)carve(512 * 4);
    f16* xsrc16   = (f16*)carve((size_t)N * 128 * 2);
    f16* xscal16  = (f16*)carve((size_t)N * 128 * 2);
    f16* q16      = (f16*)carve((size_t)N * 256 * 2);
    f16* k16      = (f16*)carve((size_t)N * 256 * 2);
    f16* v1_16    = (f16*)carve((size_t)N * 128 * 2);
    float* vf     = (float*)carve((size_t)N * 384 * 4);
    f16* v16b     = (f16*)carve((size_t)N * 384 * 2);
    float* s1     = (float*)carve((size_t)N * 96 * 4);
    float* s2     = (float*)carve((size_t)N * 96 * 4);
    float* mixed  = (float*)carve((size_t)N * 352 * 4);
    float* agg    = (float*)carve((size_t)N * 352 * 4);
    float* logits = (float*)carve((size_t)E * 8 * 4);
    float* mx     = (float*)carve((size_t)N * 8 * 4);
    float* den    = (float*)carve((size_t)N * 8 * 4);
    // f16 weights
    f16* Wq16   = (f16*)carve(256 * 128 * 2);
    f16* Wk16   = (f16*)carve(256 * 128 * 2);
    f16* Wv1_16 = (f16*)carve(128 * 128 * 2);
    f16* Wv2_16 = (f16*)carve(384 * 128 * 2);
    f16* Wrshf16= (f16*)carve(96 * 128 * 2);
    f16* Wirf16 = (f16*)carve(96 * 128 * 2);
    // transposed f16 tpw: [MO][K]
    const int pK[8]  = {4096, 2048, 2048, 1024, 4096, 2048, 2048, 1024};
    const int pMO[8] = {64, 64, 64, 64, 32, 32, 32, 32};
    f16* tpwT[8];
    for (int p = 0; p < 8; ++p) tpwT[p] = (f16*)carve((size_t)pK[p] * pMO[p] * 2);
    if (off > ws_size) return;

    // ---- Wigner-3j on host (deterministic), uploaded each call ----
    static float h_cg[512];
    {
        float* p = h_cg;
        w3j_host(1,1,1,p); p += 27;  w3j_host(1,2,1,p); p += 45;
        w3j_host(2,1,1,p); p += 45;  w3j_host(2,2,1,p); p += 75;
        w3j_host(1,1,2,p); p += 45;  w3j_host(1,2,2,p); p += 75;
        w3j_host(2,1,2,p); p += 75;  w3j_host(2,2,2,p); p += 125;
    }
    hipMemcpyAsync(d_cg, h_cg, 512 * 4, hipMemcpyHostToDevice, stream);

    const int NT = (N + 15) / 16;
    dim3 w32(32);
    auto g1 = [](long n) { return dim3((unsigned)((n + 255) / 256)); };

    // 0) weight prep (f16 / transposed f16)
    cvt_f16_kernel<<<g1(256*128), 256, 0, stream>>>(W_q,   Wq16,   256*128);
    cvt_f16_kernel<<<g1(256*128), 256, 0, stream>>>(W_k,   Wk16,   256*128);
    cvt_f16_kernel<<<g1(128*128), 256, 0, stream>>>(W_v1,  Wv1_16, 128*128);
    cvt_f16_kernel<<<g1(384*128), 256, 0, stream>>>(W_v2,  Wv2_16, 384*128);
    cvt_f16_kernel<<<g1(96*128),  256, 0, stream>>>(W_rshf,Wrshf16,96*128);
    cvt_f16_kernel<<<g1(96*128),  256, 0, stream>>>(W_irf, Wirf16, 96*128);
    cvt_f16_kernel<<<g1((long)N*128), 256, 0, stream>>>(x_scalar, xscal16, (long)N*128);
    for (int p = 0; p < 8; ++p)
        tpw_t_kernel<<<g1((long)pK[p]*pMO[p]), 256, 0, stream>>>(tpw[p], tpwT[p], pK[p], pMO[p]);

    // 1) rbf filter + x_src (f16)
    rbf_xsrc_kernel<<<g1((long)N*128), 256, 0, stream>>>(x_scalar, rbf, env, W_rbf, b_rbf, xsrc16, N);

    // 2) node GEMMs (WMMA f16, f32 accumulate)
    gemm16_kernel<<<dim3(NT,16), w32, 0, stream>>>(xsrc16, 128, 0, Wq16,   b_q,   nullptr, q16,   256, N, 128, 0);
    gemm16_kernel<<<dim3(NT,16), w32, 0, stream>>>(xscal16,128, 0, Wk16,   b_k,   nullptr, k16,   256, N, 128, 0);
    gemm16_kernel<<<dim3(NT, 8), w32, 0, stream>>>(xsrc16, 128, 0, Wv1_16, b_v1,  nullptr, v1_16, 128, N, 128, 1);
    gemm16_kernel<<<dim3(NT,24), w32, 0, stream>>>(v1_16,  128, 0, Wv2_16, b_v2,  vf,      v16b,  384, N, 128, 0);
    gemm16_kernel<<<dim3(NT, 6), w32, 0, stream>>>(v16b,   384, 0, Wrshf16,b_rshf,s1,      nullptr,96, N, 128, 0);
    gemm16_kernel<<<dim3(NT, 6), w32, 0, stream>>>(v16b,   384, 128, Wirf16, b_irf, s2,    nullptr,96, N, 128, 0);

    // 3) mixed_ireps
    etp_kernel<<<g1((long)N*352), 256, 0, stream>>>(x_vector, rsh, s1, s2, mixed, N);

    // 4) init: out scalar = residual, agg/den = 0, mx = -inf
    copyf_kernel<<<g1((long)N*128), 256, 0, stream>>>(out, x_scalar, (long)N*128);
    hipMemsetAsync(agg, 0, (size_t)N*352*4, stream);
    hipMemsetAsync(den, 0, (size_t)N*8*4, stream);
    initf_kernel<<<g1((long)N*8), 256, 0, stream>>>(mx, -1e30f, (long)N*8);

    // 5) edge attention + scatters
    att_logit_kernel<<<g1((long)E*8), 256, 0, stream>>>(q16, k16, ei, logits, mx, E);
    att_exp_kernel<<<g1((long)E*8), 256, 0, stream>>>(ei, logits, mx, den, E);
    scatter_scalar_kernel<<<g1((long)E*8), 256, 0, stream>>>(ei, logits, den, vf, out, E);
    scatter_agg_kernel<<<g1((long)E*88), 256, 0, stream>>>(ei, mixed, agg, E);

    // 6) fully-connected tensor product (WMMA) + vector residual
    fctp_kernel<0><<<NT, w32, 0, stream>>>(agg, x_vector, d_cg,
                                           tpwT[0], tpwT[1], tpwT[2], tpwT[3], out, N);
    fctp_kernel<1><<<NT, w32, 0, stream>>>(agg, x_vector, d_cg,
                                           tpwT[4], tpwT[5], tpwT[6], tpwT[7], out, N);
}
